// LIFENet_89524298318168
// MI455X (gfx1250) — matrix operand order repaired and checked by reading, not compiled, not run on hardware
//
#include <hip/hip_runtime.h>

// ---------------------------------------------------------------------------
// LIFENet forward for MI455X (gfx1250, wave32, WMMA bf16 path).
// Dims fixed by reference: B=16, T=96, D=96, K=10, KD=960, NG*D=288, FI=3, OUT=10.
// All GEMMs run on v_wmma_f32_16x16x32_bf16; B operands pre-transposed to KxN
// so B tile loads and LDS stores are b128; all strides compile-time constants.
// ---------------------------------------------------------------------------

typedef __attribute__((ext_vector_type(16))) __bf16 v16bf;
typedef __attribute__((ext_vector_type(8)))  __bf16 v8bf;
typedef __attribute__((ext_vector_type(8)))  float  v8f;

#define NB   16
#define NT   96
#define ND   96
#define NKD  960
#define N3D  288
#define NFI  3
#define NOUT 10
#define BT   (NB * NT)        // 1536

// ---------------------------------------------------------------------------
// Pack [X | M | deltas_f] -> bf16 input matrix (BT x 288)
// ---------------------------------------------------------------------------
__global__ __launch_bounds__(256) void k_pack_inp(const float* __restrict__ X,
                                                  const float* __restrict__ M,
                                                  const float* __restrict__ DF,
                                                  __bf16* __restrict__ out) {
  int i = blockIdx.x * 256 + threadIdx.x;
  if (i >= BT * N3D) return;
  int row = i / N3D, c = i % N3D;
  float v;
  if (c < ND)            v = X [row * ND + c];
  else if (c < 2 * ND)   v = M [row * ND + (c - ND)];
  else                   v = DF[row * ND + (c - 2 * ND)];
  out[i] = (__bf16)v;
}

// Convert weight (N x K f32, row-major) -> transposed bf16 (K x N row-major).
__global__ __launch_bounds__(256) void k_cvt_transpose(const float* __restrict__ in,
                                                       __bf16* __restrict__ out,
                                                       int N, int K) {
  int i = blockIdx.x * 256 + threadIdx.x;   // over K*N, out-coalesced
  if (i >= N * K) return;
  int k = i / N, n = i % N;
  out[i] = (__bf16)in[n * K + k];
}

// ---------------------------------------------------------------------------
// WMMA GEMM:  C[M,N] = A[M,K] @ Bkn   with Bkn stored K x N row-major (bf16).
//   EPI=1 : fuse  + bias[col] + posemb[t,col], ELU; also emit bf16 copy of C.
// Block = 128 threads = 4 waves; block tile 16(M) x 64(N); K stepped by 32.
// Ping-pong LDS, register-staged global loads, one barrier per K-step.
// LDA compile-time; LDB == LDC == NKD at every call site.
// ---------------------------------------------------------------------------
template <int EPI, int KDIM, int LDA>
__global__ __launch_bounds__(128) void k_gemm_bf16(
    const __bf16* __restrict__ A,   long strideA,
    const __bf16* __restrict__ Bkn, long strideB,
    float* __restrict__ C,          long strideC,
    const float* __restrict__ bias,    // EPI=1
    const float* __restrict__ posemb,  // EPI=1
    __bf16* __restrict__ Cbf) {        // EPI=1 (stride == strideC)
  constexpr int S   = KDIM / 32;
  constexpr int LDB = NKD;
  constexpr int LDC = NKD;
  __shared__ alignas(16) __bf16 shA[2][16 * 32];   // [m][k]
  __shared__ alignas(16) __bf16 shB[2][32 * 64];   // [k][n_local]

  const int tid  = threadIdx.x;
  const int wave = tid >> 5;
  const int lane = tid & 31;
  const int half = lane >> 4;      // 0: lanes 0-15, 1: lanes 16-31
  const int lm   = lane & 15;

  const int m0    = blockIdx.x * 16;
  const int n0blk = blockIdx.y * 64;
  const int bz    = blockIdx.z;

  const __bf16* Ab = A   + (long)bz * strideA;
  const __bf16* Bb = Bkn + (long)bz * strideB;
  float*        Cb = C   + (long)bz * strideC;

  // Uniform per-thread staging (no exec divergence):
  //   A tile 16x32 = 128 chunks of 4 bf16 (uint2, 8 B) -> 1 per thread.
  //   B tile 32x64 = 256 chunks of 8 bf16 (uint4, 16 B) -> 2 per thread.
  const int ar = tid >> 3, aq = (tid & 7) << 2;
  const int br0 = tid >> 3,         bq0 = (tid & 7) << 3;
  const int br1 = (tid + 128) >> 3, bq1 = bq0;

  uint2 rA;
  uint4 rB0, rB1;
  // preload tile 0 into registers, stage into buffer 0
  rA  = *(const uint2*)&Ab[(long)(m0 + ar) * LDA + aq];
  rB0 = *(const uint4*)&Bb[(long)br0 * LDB + n0blk + bq0];
  rB1 = *(const uint4*)&Bb[(long)br1 * LDB + n0blk + bq1];
  *(uint2*)&shA[0][ar * 32 + aq]   = rA;
  *(uint4*)&shB[0][br0 * 64 + bq0] = rB0;
  *(uint4*)&shB[0][br1 * 64 + bq1] = rB1;

  const int kb = (half << 4) + lm;        // this lane's K index within the tile
  v8f acc = {};

#pragma unroll
  for (int s = 0; s < S; ++s) {
    // issue next tile's global loads (latency hidden behind WMMA below)
    if (s + 1 < S) {
      const int k0 = (s + 1) * 32;
      rA  = *(const uint2*)&Ab[(long)(m0 + ar) * LDA + k0 + aq];
      rB0 = *(const uint4*)&Bb[(long)(k0 + br0) * LDB + n0blk + bq0];
      rB1 = *(const uint4*)&Bb[(long)(k0 + br1) * LDB + n0blk + bq1];
    }
    __syncthreads();   // buffer (s&1) fully staged

    const int cur = s & 1;
    // A fragment: lane=row lm; elements 0-7 -> K=half*8+(0..7), 8-15 -> K=16+half*8+(0..7)
    v8bf a0 = *(const v8bf*)&shA[cur][lm * 32 + (half << 3)];
    v8bf a1 = *(const v8bf*)&shA[cur][lm * 32 + 16 + (half << 3)];
    // B fragment: lane col N = wave*16 + lm; elements 0-7 -> K=half*8+(0..7), 8-15 -> K=16+half*8+(0..7)
    v16bf fa = __builtin_shufflevector(a0, a1, 0, 1, 2, 3, 4, 5, 6, 7,
                                       8, 9, 10, 11, 12, 13, 14, 15);
    v16bf fb;
#pragma unroll
    for (int e = 0; e < 8; ++e) {
      fb[e]     = shB[cur][((half << 3) + e) * 64 + (wave << 4) + lm];
      fb[8 + e] = shB[cur][(16 + (half << 3) + e) * 64 + (wave << 4) + lm];
    }
    acc = __builtin_amdgcn_wmma_f32_16x16x32_bf16(
        /*neg_a=*/false, fa, /*neg_b=*/false, fb,
        /*c_mod=*/(short)0, acc, /*reuse_a=*/false, /*reuse_b=*/false);

    // stage next tile into the other buffer (reads of it wait on next barrier)
    if (s + 1 < S) {
      const int nxt = (s + 1) & 1;
      *(uint2*)&shA[nxt][ar * 32 + aq]   = rA;
      *(uint4*)&shB[nxt][br0 * 64 + bq0] = rB0;
      *(uint4*)&shB[nxt][br1 * 64 + bq1] = rB1;
    }
  }

  // C/D layout: VGPR r, lane l -> row m0 + (l/16)*8 + r, col n0 + wave*16 + (l%16)
  const int rowb = m0 + (half << 3);
  const int col  = n0blk + (wave << 4) + lm;
#pragma unroll
  for (int r = 0; r < 8; ++r) {
    float v = acc[r];
    const int row = rowb + r;
    if (EPI == 1) {
      const int t = row % NT;
      v += bias[col] + posemb[t * NKD + col];
      v = (v > 0.0f) ? v : (__expf(v) - 1.0f);           // ELU
      Cb[(long)row * LDC + col] = v;
      (Cbf + (long)bz * strideC)[(long)row * LDC + col] = (__bf16)v;
    } else {
      Cb[(long)row * LDC + col] = v;
    }
  }
}

// ---------------------------------------------------------------------------
// Additive attention logits + softmax.  One block per (b,i).  8 waves:
//   logits[j] = sum_d tanh(key[b,j,d] + qry[b,i,d]) * w_a[d]  (wave per j)
// Softmax over j -> scores stored bf16 for the batched h_a WMMA GEMM.
// ---------------------------------------------------------------------------
__global__ __launch_bounds__(256) void k_attn_scores(const float* __restrict__ key,
                                                     const float* __restrict__ qry,
                                                     const float* __restrict__ wa,
                                                     __bf16* __restrict__ scores_bf) {
  __shared__ float q_sh[NKD];
  __shared__ float log_sh[NT];
  __shared__ float denom_sh;

  const int bi   = blockIdx.x;           // b*T + i
  const int b    = bi / NT;
  const int tid  = threadIdx.x;
  const int lane = tid & 31;
  const int wave = tid >> 5;

  const float* qrow  = qry + (long)bi * NKD;
  const float* kbase = key + (long)b * NT * NKD;

  for (int d = tid; d < NKD; d += 256) q_sh[d] = qrow[d];
  __syncthreads();

  for (int j = wave; j < NT; j += 8) {
    const float* krow = kbase + (long)j * NKD;
    float p = 0.0f;
    for (int d = lane; d < NKD; d += 32)
      p += tanhf(krow[d] + q_sh[d]) * wa[d];
    for (int off = 16; off > 0; off >>= 1)
      p += __shfl_down(p, off, 32);
    if (lane == 0) log_sh[j] = p;
  }
  __syncthreads();

  if (tid < 32) {  // wave 0: softmax normalizers over 96 logits
    float m = -3.0e30f;
    for (int j = lane; j < NT; j += 32) m = fmaxf(m, log_sh[j]);
    for (int off = 16; off > 0; off >>= 1) m = fmaxf(m, __shfl_down(m, off, 32));
    m = __shfl(m, 0, 32);
    float s = 0.0f;
    for (int j = lane; j < NT; j += 32) {
      float e = __expf(log_sh[j] - m);
      log_sh[j] = e;
      s += e;
    }
    for (int off = 16; off > 0; off >>= 1) s += __shfl_down(s, off, 32);
    if (lane == 0) denom_sh = s;
  }
  __syncthreads();

  const float inv = 1.0f / denom_sh;
  if (tid < NT) scores_bf[(long)bi * NT + tid] = (__bf16)(log_sh[tid] * inv);
}

// ---------------------------------------------------------------------------
// Temporal decay: h_r = m*h_a + (1-m)*(gamma*h_fwd + (1-gamma)*h_a)
// ---------------------------------------------------------------------------
__global__ __launch_bounds__(256) void k_temporal_decay(const float* __restrict__ Mmask,
                                                        const float* __restrict__ DF,
                                                        const float* __restrict__ Wtd,
                                                        const float* __restrict__ btd,
                                                        const float* __restrict__ ha,
                                                        float* __restrict__ hr) {
  long i = (long)blockIdx.x * 256 + threadIdx.x;       // over B*T*KD
  if (i >= (long)BT * NKD) return;
  const int  c  = (int)(i % NKD);
  const long bt = i / NKD;
  const int  t  = (int)(bt % NT);
  const int  d  = c % ND;

  const float m  = Mmask[bt * ND + d];
  const float df = DF[bt * ND + d];
  const float g  = __expf(-fmaxf(0.0f, df * Wtd[c] + btd[c]));
  const int   tp = t - ((int)df - 1);                   // valid: df in [1, t+1]
  const float hc = ha[i];
  const float hf = ha[(bt - t + tp) * NKD + c];
  hr[i] = m * hc + (1.0f - m) * (g * hf + (1.0f - g) * hc);
}

// ---------------------------------------------------------------------------
// Group-diagonal imputation: imput[b,t,d] = sum_k hr[b,t,k*D+d]*W_imp[d,k*D+d] + b_imp[d]
// ---------------------------------------------------------------------------
__global__ __launch_bounds__(256) void k_imputation(const float* __restrict__ hr,
                                                    const float* __restrict__ Wimp,
                                                    const float* __restrict__ bimp,
                                                    float* __restrict__ out) {
  int i = blockIdx.x * 256 + threadIdx.x;              // over B*T*D
  if (i >= BT * ND) return;
  const int  d  = i % ND;
  const long bt = i / ND;
  const float* h = hr + bt * NKD;
  float s = bimp[d];
#pragma unroll
  for (int k = 0; k < 10; ++k)
    s += h[k * ND + d] * Wimp[(long)d * NKD + k * ND + d];
  out[i] = s;
}

// ---------------------------------------------------------------------------
// Dense interpolation + final FC.  One block per batch b.
//   v[d,f] = sum_t Wi[f,t] * hr[b,t,d];  out[b,o] = v_flat . W_fc[o] + b_fc[o]
// ---------------------------------------------------------------------------
__global__ __launch_bounds__(256) void k_interp_fc(const float* __restrict__ hr,
                                                   const float* __restrict__ Wfc,
                                                   const float* __restrict__ bfc,
                                                   float* __restrict__ out) {
  __shared__ float v_sh[NKD * NFI];                    // 2880 floats
  const int b   = blockIdx.x;
  const int tid = threadIdx.x;
  const float* h = hr + (long)b * NT * NKD;

  for (int idx = tid; idx < NKD * NFI; idx += 256) {
    const int d = idx / NFI, f = idx % NFI;
    float s = 0.0f;
    for (int t = 0; t < NT; ++t) {
      const float p = (float)NFI * (t + 1.0f) / (float)NT;
      float w = 1.0f - fabsf(p - (1.0f + f)) / (float)NFI;
      s += w * w * h[(long)t * NKD + d];
    }
    v_sh[idx] = s;
  }
  __syncthreads();

  const int lane = tid & 31, wave = tid >> 5;
  for (int o = wave; o < NOUT; o += 8) {
    const float* wr = Wfc + (long)o * (NKD * NFI);
    float p = 0.0f;
    for (int c = lane; c < NKD * NFI; c += 32) p += wr[c] * v_sh[c];
    for (int off = 16; off > 0; off >>= 1) p += __shfl_down(p, off, 32);
    if (lane == 0) out[b * NOUT + o] = p + bfc[o];
  }
}

// ---------------------------------------------------------------------------
// Host-side orchestration
// ---------------------------------------------------------------------------
extern "C" void kernel_launch(void* const* d_in, const int* in_sizes, int n_in,
                              void* d_out, int out_size, void* d_ws, size_t ws_size,
                              hipStream_t stream) {
  (void)in_sizes; (void)n_in; (void)out_size; (void)ws_size;
  const float* X      = (const float*)d_in[0];
  const float* Mmask  = (const float*)d_in[1];
  const float* DF     = (const float*)d_in[2];
  const float* W_e    = (const float*)d_in[3];
  const float* b_e    = (const float*)d_in[4];
  const float* posemb = (const float*)d_in[5];
  const float* W_a    = (const float*)d_in[6];
  const float* V_a    = (const float*)d_in[7];
  const float* w_a    = (const float*)d_in[8];
  const float* W_td   = (const float*)d_in[9];
  const float* b_td   = (const float*)d_in[10];
  const float* W_imp  = (const float*)d_in[11];
  const float* b_imp  = (const float*)d_in[12];
  const float* W_fc   = (const float*)d_in[13];
  const float* b_fc   = (const float*)d_in[14];
  float* out = (float*)d_out;                  // [0,160) = out, [160, ...) = imput

  // ---- carve workspace ----
  char* w = (char*)d_ws;
  auto carve = [&](size_t bytes) -> void* {
    void* p = (void*)w;
    w += (bytes + 255) & ~(size_t)255;
    return p;
  };
  __bf16* inp_bf    = (__bf16*)carve((size_t)BT * N3D * 2);
  __bf16* WeT_bf    = (__bf16*)carve((size_t)N3D * NKD * 2);   // 288 x 960 (KxN)
  __bf16* WaT_bf    = (__bf16*)carve((size_t)NKD * NKD * 2);   // 960 x 960 (KxN)
  __bf16* VaT_bf    = (__bf16*)carve((size_t)NKD * NKD * 2);
  __bf16* he_bf     = (__bf16*)carve((size_t)BT * NKD * 2);    // per-batch 96x960 = KxN
  float*  he32      = (float*) carve((size_t)BT * NKD * 4);
  float*  key32     = (float*) carve((size_t)BT * NKD * 4);
  float*  qry32     = (float*) carve((size_t)BT * NKD * 4);
  __bf16* scores_bf = (__bf16*)carve((size_t)NB * NT * NT * 2);
  float*  ha32      = (float*) carve((size_t)BT * NKD * 4);
  float*  hr32      = (float*) carve((size_t)BT * NKD * 4);

  // ---- 1: pack inputs; convert+transpose weights to KxN bf16 ----
  k_pack_inp<<<(BT * N3D + 255) / 256, 256, 0, stream>>>(X, Mmask, DF, inp_bf);
  k_cvt_transpose<<<(NKD * N3D + 255) / 256, 256, 0, stream>>>(W_e, WeT_bf, NKD, N3D);
  k_cvt_transpose<<<(NKD * NKD + 255) / 256, 256, 0, stream>>>(W_a, WaT_bf, NKD, NKD);
  k_cvt_transpose<<<(NKD * NKD + 255) / 256, 256, 0, stream>>>(V_a, VaT_bf, NKD, NKD);

  // ---- 2: h_e = ELU(inp @ W_e^T + b_e + pos_emb)   (WMMA, fused epilogue) ----
  k_gemm_bf16<1, N3D, N3D><<<dim3(BT / 16, NKD / 64, 1), 128, 0, stream>>>(
      inp_bf, 0, WeT_bf, 0, he32, 0, b_e, posemb, he_bf);

  // ---- 3: key_v = h_e @ W_a^T ; qry_v = h_e @ V_a^T   (WMMA) ----
  k_gemm_bf16<0, NKD, NKD><<<dim3(BT / 16, NKD / 64, 1), 128, 0, stream>>>(
      he_bf, 0, WaT_bf, 0, key32, 0, nullptr, nullptr, nullptr);
  k_gemm_bf16<0, NKD, NKD><<<dim3(BT / 16, NKD / 64, 1), 128, 0, stream>>>(
      he_bf, 0, VaT_bf, 0, qry32, 0, nullptr, nullptr, nullptr);

  // ---- 4: additive-attention logits + softmax (tanh-bound, VALU) ----
  k_attn_scores<<<BT, 256, 0, stream>>>(key32, qry32, w_a, scores_bf);

  // ---- 5: h_a = scores @ h_e   (batched WMMA; h_e per batch is already KxN) ----
  k_gemm_bf16<0, NT, NT><<<dim3(NT / 16, NKD / 64, NB), 128, 0, stream>>>(
      scores_bf, (long)NT * NT,
      he_bf, (long)NT * NKD,
      ha32, (long)NT * NKD,
      nullptr, nullptr, nullptr);

  // ---- 6: temporal decay ----
  k_temporal_decay<<<(int)(((long)BT * NKD + 255) / 256), 256, 0, stream>>>(
      Mmask, DF, W_td, b_td, ha32, hr32);

  // ---- 7: imputation -> d_out[160:] ----
  k_imputation<<<(BT * ND + 255) / 256, 256, 0, stream>>>(hr32, W_imp, b_imp,
                                                          out + NB * NOUT);

  // ---- 8: dense interpolation + FC -> d_out[:160] ----
  k_interp_fc<<<NB, 256, 0, stream>>>(hr32, W_fc, b_fc, out);
}
